// LearningHiddenMarkovModelFilterBlock_10282151706645
// MI455X (gfx1250) — matrix operand order repaired and checked by reading, not compiled, not run on hardware
//
#include <hip/hip_runtime.h>
#include <hip/hip_bf16.h>

// ---------------------------------------------------------------------------
// HMM filter: B=128, T=512, S=256  (matches reference setup_inputs)
//   transition = softmax(weight, axis=1)      [S,S]  (loop-invariant)
//   state0     = softmax(initial_state)       [S]
//   per step:  unnorm = state * emission[:,t,:]; state = unnorm / max(L1,eps)
//              next   = state @ transition;    out[:,t,:] = next
// Strategy: 8 workgroups x 256 threads (8 waves). Each WG owns 16 batch rows
// for the full sequential T loop. Transition held as f16 WMMA B-fragments in
// registers for the whole loop; A-fragments built per step in LDS; f32 accum.
// 4 independent WMMA accumulator chains to hide XDL pipeline latency.
// ---------------------------------------------------------------------------

typedef __attribute__((ext_vector_type(16))) _Float16 v16h;
typedef __attribute__((ext_vector_type(8)))  float    v8f;

#define HMM_B   128
#define HMM_T   512
#define HMM_S   256
#define BT      16            // batch tile per workgroup
#define LDH     264           // padded half stride  (264*2 = 528B, 16B aligned, bank-staggered)
#define LDF     264           // padded float stride (264*4 = 1056B, 16B aligned, bank-staggered)
#define EPSF    1e-12f

// ---------------- prep: row softmax of weight -> f16 transition -------------
__global__ void hmm_prep_transition(const float* __restrict__ W,
                                    _Float16* __restrict__ trans) {
    __shared__ float red[HMM_S];
    const int row = blockIdx.x;
    const int tid = threadIdx.x;               // 256 threads = 256 cols
    float v = W[row * HMM_S + tid];
    red[tid] = v; __syncthreads();
    for (int s = 128; s > 0; s >>= 1) {
        if (tid < s) red[tid] = fmaxf(red[tid], red[tid + s]);
        __syncthreads();
    }
    float m = red[0]; __syncthreads();
    float e = __expf(v - m);
    red[tid] = e; __syncthreads();
    for (int s = 128; s > 0; s >>= 1) {
        if (tid < s) red[tid] += red[tid + s];
        __syncthreads();
    }
    trans[row * HMM_S + tid] = (_Float16)(e / red[0]);
}

// ---------------- prep: softmax of initial_state ----------------------------
__global__ void hmm_prep_init(const float* __restrict__ x, float* __restrict__ y) {
    __shared__ float red[HMM_S];
    const int tid = threadIdx.x;
    float v = x[tid];
    red[tid] = v; __syncthreads();
    for (int s = 128; s > 0; s >>= 1) {
        if (tid < s) red[tid] = fmaxf(red[tid], red[tid + s]);
        __syncthreads();
    }
    float m = red[0]; __syncthreads();
    float e = __expf(v - m);
    red[tid] = e; __syncthreads();
    for (int s = 128; s > 0; s >>= 1) {
        if (tid < s) red[tid] += red[tid + s];
        __syncthreads();
    }
    y[tid] = e / red[0];
}

// ---------------- main sequential filter ------------------------------------
__global__ __launch_bounds__(256, 1)
void hmm_filter(const float* __restrict__ em,       // [B,T,S]
                const _Float16* __restrict__ trans, // [S,S] row-major (K,N)
                const float* __restrict__ init,     // [S] softmaxed
                float* __restrict__ out) {          // [B,T,S]
    __shared__ __align__(16) float    sF32[BT * LDF];   // f32 state (next_state)
    __shared__ __align__(16) _Float16 sF16[BT * LDH];   // f16 normalized state (A matrix)

    const int tid  = threadIdx.x;
    const int r    = tid >> 4;          // batch row within tile, 0..15
    const int c    = tid & 15;          // 16-col chunk within S,  0..15
    const int lane = tid & 31;          // wave32 lane
    const int w    = tid >> 5;          // wave id 0..7
    const int nlo  = lane & 15;
    const int b0   = blockIdx.x * BT;

    // ---- load this wave's B-fragments (transition) into registers, once ----
    // B fragments (contiguous k order, same as A): lanes 0-15 hold K=0..15, lanes 16-31 K=16..31,
    // column N = tileBase + (lane&15); element e of v16h = K kbase+e.
    const int bkoff = (lane & 16);      // 0 or 16
    v16h bfrag[2][8];
#pragma unroll
    for (int jj = 0; jj < 2; ++jj) {
        const int n = w * 32 + jj * 16 + nlo;
#pragma unroll
        for (int kt = 0; kt < 8; ++kt) {
            const int kb = kt * 32 + bkoff;
            v16h b;
#pragma unroll
            for (int e = 0; e < 16; ++e)
                b[e] = trans[(kb + e) * HMM_S + n];
            bfrag[jj][kt] = b;
        }
    }

    // ---- initial belief broadcast into f32 state ----
#pragma unroll
    for (int i = 0; i < 16; ++i)
        sF32[r * LDF + c * 16 + i] = init[c * 16 + i];
    __syncthreads();

    // ---- emission pointers + first-step buffer ----
    const float* emRow = em + ((size_t)(b0 + r) * HMM_T) * HMM_S + (size_t)c * 16;
    float4 emBuf[4];
#pragma unroll
    for (int q = 0; q < 4; ++q) emBuf[q] = ((const float4*)emRow)[q];

    const int akoff  = (lane & 16);             // A chunk offset: 0 / 16
    const int rowOff = (lane & 16) ? 8 : 0;     // C/D row offset

    for (int t = 0; t < HMM_T; ++t) {
        // ---- Bayes update: unnorm = state*em, L1 normalize, round to f16 ----
        float u[16];
        float part = 0.f;
#pragma unroll
        for (int q = 0; q < 4; ++q) {
            const float4 e4 = emBuf[q];
            const float4 s4 = *(const float4*)&sF32[r * LDF + c * 16 + q * 4];
            u[q * 4 + 0] = s4.x * e4.x;  part += fabsf(u[q * 4 + 0]);
            u[q * 4 + 1] = s4.y * e4.y;  part += fabsf(u[q * 4 + 1]);
            u[q * 4 + 2] = s4.z * e4.z;  part += fabsf(u[q * 4 + 2]);
            u[q * 4 + 3] = s4.w * e4.w;  part += fabsf(u[q * 4 + 3]);
        }
        // 16 threads of one batch row == one half-wave: xor-reduce, no barrier
        part += __shfl_xor(part, 1, 32);
        part += __shfl_xor(part, 2, 32);
        part += __shfl_xor(part, 4, 32);
        part += __shfl_xor(part, 8, 32);
        const float scale = 1.f / fmaxf(part, EPSF);
#pragma unroll
        for (int i = 0; i < 16; ++i)
            sF16[r * LDH + c * 16 + i] = (_Float16)(u[i] * scale);
        __syncthreads();   // sF16 visible to all waves; sF32 reads done

        // ---- prefetch next step's emissions while WMMA chain runs ----
        float4 emNext[4] = {};
        if (t + 1 < HMM_T) {
            const float4* ep = (const float4*)(emRow + (size_t)(t + 1) * HMM_S);
#pragma unroll
            for (int q = 0; q < 4; ++q) emNext[q] = ep[q];
        }
        if (t + 2 < HMM_T)
            __builtin_prefetch(emRow + (size_t)(t + 2) * HMM_S, 0, 1);

        // ---- next = stateF16 @ transition : 16 WMMAs per wave -------------
        // A fragments (contiguous k order, same as B): row M = lane&15; lanes0-15 chunks K+0/K+8,
        // lanes16-31 chunks K+16/K+24; each 8-half chunk is one ds_load_b128.
        // 4 independent accumulator chains (split K in half) so dependent
        // WMMAs are 4 XDL slots apart -> pipeline stays full.
        v8f accA0 = {}, accA1 = {};   // N-tile 0: even/odd kt chains
        v8f accB0 = {}, accB1 = {};   // N-tile 1: even/odd kt chains
#pragma unroll
        for (int kp = 0; kp < 4; ++kp) {
            const int kt0 = kp * 2;
            const int kt1 = kp * 2 + 1;
            const _Float16* ap0 = &sF16[nlo * LDH + kt0 * 32 + akoff];
            const _Float16* ap1 = &sF16[nlo * LDH + kt1 * 32 + akoff];
            union { uint4 q[2]; v16h h; } av0, av1;
            av0.q[0] = *(const uint4*)(ap0);
            av0.q[1] = *(const uint4*)(ap0 + 8);
            av1.q[0] = *(const uint4*)(ap1);
            av1.q[1] = *(const uint4*)(ap1 + 8);
            accA0 = __builtin_amdgcn_wmma_f32_16x16x32_f16(
                false, av0.h, false, bfrag[0][kt0], (short)0, accA0, false, false);
            accB0 = __builtin_amdgcn_wmma_f32_16x16x32_f16(
                false, av0.h, false, bfrag[1][kt0], (short)0, accB0, false, false);
            accA1 = __builtin_amdgcn_wmma_f32_16x16x32_f16(
                false, av1.h, false, bfrag[0][kt1], (short)0, accA1, false, false);
            accB1 = __builtin_amdgcn_wmma_f32_16x16x32_f16(
                false, av1.h, false, bfrag[1][kt1], (short)0, accB1, false, false);
        }

        // ---- combine chains + scatter D fragments back to f32 state ----
        {
            const int n0 = w * 32 + nlo;
#pragma unroll
            for (int i = 0; i < 8; ++i) {
                sF32[(i + rowOff) * LDF + n0]      = accA0[i] + accA1[i];
                sF32[(i + rowOff) * LDF + n0 + 16] = accB0[i] + accB1[i];
            }
        }
        __syncthreads();   // sF32 = next_state, visible

        // ---- coalesced trajectory write: out[:,t,:] = next_state ----
        float* op = out + ((size_t)(b0 + r) * HMM_T + t) * HMM_S + (size_t)c * 16;
#pragma unroll
        for (int q = 0; q < 4; ++q)
            ((float4*)op)[q] = *(const float4*)&sF32[r * LDF + c * 16 + q * 4];

#pragma unroll
        for (int q = 0; q < 4; ++q) emBuf[q] = emNext[q];
    }
}

// ---------------------------------------------------------------------------
extern "C" void kernel_launch(void* const* d_in, const int* in_sizes, int n_in,
                              void* d_out, int out_size, void* d_ws, size_t ws_size,
                              hipStream_t stream) {
    (void)in_sizes; (void)n_in; (void)out_size; (void)ws_size;
    const float* em   = (const float*)d_in[0];  // [B,T,S]
    const float* W    = (const float*)d_in[1];  // [S,S]
    const float* init = (const float*)d_in[2];  // [S]
    float* out        = (float*)d_out;

    // workspace: [0, S*S halves) f16 transition; then S floats softmax(init)
    _Float16* trans   = (_Float16*)d_ws;
    float*    initSft = (float*)((char*)d_ws + (size_t)HMM_S * HMM_S * sizeof(_Float16));

    hmm_prep_transition<<<HMM_S, HMM_S, 0, stream>>>(W, trans);
    hmm_prep_init<<<1, HMM_S, 0, stream>>>(init, initSft);
    hmm_filter<<<HMM_B / BT, 256, 0, stream>>>(em, trans, initSft, out);
}
